// OmniBlock_89721866813909
// MI455X (gfx1250) — compile-verified, matrix operand order checked by reading, not run on hardware
//
#include <hip/hip_runtime.h>
#include <hip/hip_bf16.h>

// ---------------------------------------------------------------------------
// Types for CDNA5 WMMA (wave32): V_WMMA_F32_16X16X32_BF16
// ---------------------------------------------------------------------------
typedef __attribute__((ext_vector_type(16))) __bf16 v16bf;
typedef __attribute__((ext_vector_type(8)))  __bf16 v8bf;
typedef __attribute__((ext_vector_type(8)))  float  v8f;

#define BB 2
#define TT 2048
#define CC 1024
#define HH 16
#define DD 64
#define EE 8
#define HDIM 1536
#define MM (BB*TT)

__device__ __forceinline__ v8f vzero8() {
  v8f z = {0.f,0.f,0.f,0.f,0.f,0.f,0.f,0.f};
  return z;
}

__device__ __forceinline__ v16bf frag_combine(v8bf lo, v8bf hi) {
  return __builtin_shufflevector(lo, hi, 0,1,2,3,4,5,6,7,8,9,10,11,12,13,14,15);
}

// Load a 16x32 bf16 A/B fragment from LDS. Row-major [row][k], `stride` in
// elements (layout per ISA 7.12.2: lanes 0-15 K={0..7,16..23},
// lanes 16-31 K={8..15,24..31}).
__device__ __forceinline__ v16bf load_frag(const __bf16* base, int stride, int lane) {
  int r  = lane & 15;
  int g8 = (lane >> 4) * 8;
  const __bf16* p = base + (size_t)r * stride;
  v8bf lo = *(const v8bf*)(p + g8);
  v8bf hi = *(const v8bf*)(p + 16 + g8);
  return frag_combine(lo, hi);
}

__device__ __forceinline__ v8f wmma_bf16(v16bf a, v16bf b, v8f c) {
  return __builtin_amdgcn_wmma_f32_16x16x32_bf16(false, a, false, b, (short)0, c, false, false);
}

// ---------------------------------------------------------------------------
// CDNA5 async global->LDS copy (GVS mode: uniform 64-bit SGPR base +
// per-lane 32-bit byte offset; VDST holds the per-lane LDS byte address).
// Tracked by ASYNCcnt.
// ---------------------------------------------------------------------------
__device__ __forceinline__ void async_b128(void* lds_ptr, unsigned gbyte_off,
                                           const void* sbase) {
  unsigned lds_addr = (unsigned)(uintptr_t)lds_ptr;   // low 32 bits = LDS offset
  asm volatile("global_load_async_to_lds_b128 %0, %1, %2 offset:0"
               :: "v"(lds_addr), "v"(gbyte_off), "s"(sbase)
               : "memory");
}
__device__ __forceinline__ void wait_async0() {
  asm volatile("s_wait_asynccnt 0" ::: "memory");
}

__device__ __forceinline__ float fast_rcp(float x) {
  return __builtin_amdgcn_rcpf(x);
}

// ---------------------------------------------------------------------------
// Elementwise fp32 -> bf16 cast (weights)
// ---------------------------------------------------------------------------
__global__ void castf2b(const float* __restrict__ in, __bf16* __restrict__ out, size_t n) {
  size_t i = (size_t)blockIdx.x * blockDim.x + threadIdx.x;
  size_t stride = (size_t)gridDim.x * blockDim.x;
  for (; i < n; i += stride) out[i] = (__bf16)in[i];
}

// ---------------------------------------------------------------------------
// RMSNorm + cast to bf16. One block (256 thr) per token row of length CC.
// ---------------------------------------------------------------------------
__global__ void rmsnorm_cast(const float* __restrict__ x, const float* __restrict__ w,
                             __bf16* __restrict__ out) {
  __shared__ float red[256];
  int t = blockIdx.x;
  int tid = threadIdx.x;
  const float* xr = x + (size_t)t * CC;
  float ss = 0.f;
  for (int c = tid; c < CC; c += 256) { float v = xr[c]; ss += v * v; }
  red[tid] = ss;
  __syncthreads();
  for (int s = 128; s > 0; s >>= 1) {
    if (tid < s) red[tid] += red[tid + s];
    __syncthreads();
  }
  float rs = rsqrtf(red[0] * (1.0f / CC) + 1e-6f);
  for (int c = tid; c < CC; c += 256)
    out[(size_t)t * CC + c] = (__bf16)(w[c] * xr[c] * rs);
}

// ---------------------------------------------------------------------------
// Generic tiled WMMA GEMM:  out[M,N] = A[M,K] * W[N,K]^T
// Tile 128(M) x 64(N), K-step 64, double-buffered LDS filled with
// global_load_async_to_lds_b128.  8 waves: 4(M) x 2(N), each wave 32x32.
// MODE 0: store bf16          (QKV projections)
// MODE 1: f32 = acc + resid, also mirrored to outf2  (O-proj + residual)
// MODE 2: outf += gate[row]*acc                       (MoE down-proj)
// ---------------------------------------------------------------------------
template<int MODE>
__global__ void gemm_wmma(const __bf16* __restrict__ A, const __bf16* __restrict__ W,
                          int M, int N, int K,
                          __bf16* __restrict__ outb,
                          float* __restrict__ outf, float* __restrict__ outf2,
                          const float* __restrict__ resid,
                          const float* __restrict__ gate, int gcol) {
  __shared__ __align__(16) __bf16 As[2][128][72];   // 64 + 8 pad -> 144B rows
  __shared__ __align__(16) __bf16 Bs[2][64][72];

  int tid  = threadIdx.x;
  int lane = tid & 31;
  int wave = tid >> 5;
  int wm = wave >> 1;        // 0..3
  int wn = wave & 1;         // 0..1
  int m0 = blockIdx.y * 128;
  int n0 = blockIdx.x * 64;

  const __bf16* Abase = A + (size_t)m0 * K;
  const __bf16* Wbase = W + (size_t)n0 * K;

  auto stage = [&](int buf, int k0) {
    const void* ab = (const void*)(Abase + k0);
    #pragma unroll
    for (int t = 0; t < 4; ++t) {                 // A: 128x64 = 1024 16B chunks
      int c = tid + t * 256;
      int row = c >> 3, koff = (c & 7) * 8;
      async_b128(&As[buf][row][koff], (unsigned)((row * K + koff) * 2), ab);
    }
    const void* bb = (const void*)(Wbase + k0);
    #pragma unroll
    for (int t = 0; t < 2; ++t) {                 // B: 64x64 = 512 16B chunks
      int c = tid + t * 256;
      int row = c >> 3, koff = (c & 7) * 8;
      async_b128(&Bs[buf][row][koff], (unsigned)((row * K + koff) * 2), bb);
    }
  };

  v8f acc[2][2];
  acc[0][0] = vzero8(); acc[0][1] = vzero8();
  acc[1][0] = vzero8(); acc[1][1] = vzero8();

  stage(0, 0);
  wait_async0();
  __syncthreads();

  int cur = 0;
  for (int k0 = 0; k0 < K; k0 += 64) {
    if (k0 + 64 < K) stage(cur ^ 1, k0 + 64);
    #pragma unroll
    for (int kk = 0; kk < 2; ++kk) {
      v16bf af[2], bf[2];
      #pragma unroll
      for (int ms = 0; ms < 2; ++ms)
        af[ms] = load_frag(&As[cur][wm*32 + ms*16][kk*32], 72, lane);
      #pragma unroll
      for (int ns = 0; ns < 2; ++ns)
        bf[ns] = load_frag(&Bs[cur][wn*32 + ns*16][kk*32], 72, lane);
      #pragma unroll
      for (int ms = 0; ms < 2; ++ms)
        #pragma unroll
        for (int ns = 0; ns < 2; ++ns)
          acc[ms][ns] = wmma_bf16(af[ms], bf[ns], acc[ms][ns]);
    }
    wait_async0();
    __syncthreads();
    cur ^= 1;
  }

  // Epilogue.  C layout: lane&15 = column, vgpr r = row (+8 for lanes 16-31).
  #pragma unroll
  for (int ms = 0; ms < 2; ++ms) {
    #pragma unroll
    for (int ns = 0; ns < 2; ++ns) {
      int rbase = m0 + wm*32 + ms*16 + ((lane >> 4) * 8);
      int col   = n0 + wn*32 + ns*16 + (lane & 15);
      #pragma unroll
      for (int r = 0; r < 8; ++r) {
        size_t idx = (size_t)(rbase + r) * N + col;
        float v = acc[ms][ns][r];
        if (MODE == 0) {
          outb[idx] = (__bf16)v;
        } else if (MODE == 1) {
          float o = v + resid[idx];
          outf[idx] = o;
          outf2[idx] = o;
        } else {
          outf[idx] += gate[(size_t)(rbase + r) * EE + gcol] * v;
        }
      }
    }
  }
}

// ---------------------------------------------------------------------------
// MoE up-proj fused:  G = silu(A*W1^T) * (A*W3^T), bf16 out. Same pipeline,
// two B tiles and two accumulator sets (16 WMMAs per buffer swap).
// ---------------------------------------------------------------------------
__global__ void moe13_wmma(const __bf16* __restrict__ A,
                           const __bf16* __restrict__ W1, const __bf16* __restrict__ W3,
                           __bf16* __restrict__ G, int K, int N) {
  __shared__ __align__(16) __bf16 As[2][128][72];
  __shared__ __align__(16) __bf16 B1[2][64][72];
  __shared__ __align__(16) __bf16 B3[2][64][72];

  int tid  = threadIdx.x;
  int lane = tid & 31;
  int wave = tid >> 5;
  int wm = wave >> 1, wn = wave & 1;
  int m0 = blockIdx.y * 128, n0 = blockIdx.x * 64;

  const __bf16* Abase  = A  + (size_t)m0 * K;
  const __bf16* W1base = W1 + (size_t)n0 * K;
  const __bf16* W3base = W3 + (size_t)n0 * K;

  auto stage = [&](int buf, int k0) {
    const void* ab = (const void*)(Abase + k0);
    #pragma unroll
    for (int t = 0; t < 4; ++t) {
      int c = tid + t * 256;
      int row = c >> 3, koff = (c & 7) * 8;
      async_b128(&As[buf][row][koff], (unsigned)((row * K + koff) * 2), ab);
    }
    const void* b1 = (const void*)(W1base + k0);
    const void* b3 = (const void*)(W3base + k0);
    #pragma unroll
    for (int t = 0; t < 2; ++t) {
      int c = tid + t * 256;
      int row = c >> 3, koff = (c & 7) * 8;
      async_b128(&B1[buf][row][koff], (unsigned)((row * K + koff) * 2), b1);
      async_b128(&B3[buf][row][koff], (unsigned)((row * K + koff) * 2), b3);
    }
  };

  v8f a1[2][2], a3[2][2];
  #pragma unroll
  for (int i = 0; i < 2; ++i)
    #pragma unroll
    for (int j = 0; j < 2; ++j) { a1[i][j] = vzero8(); a3[i][j] = vzero8(); }

  stage(0, 0);
  wait_async0();
  __syncthreads();

  int cur = 0;
  for (int k0 = 0; k0 < K; k0 += 64) {
    if (k0 + 64 < K) stage(cur ^ 1, k0 + 64);
    #pragma unroll
    for (int kk = 0; kk < 2; ++kk) {
      v16bf af[2];
      #pragma unroll
      for (int ms = 0; ms < 2; ++ms)
        af[ms] = load_frag(&As[cur][wm*32 + ms*16][kk*32], 72, lane);
      #pragma unroll
      for (int ns = 0; ns < 2; ++ns) {
        v16bf b1f = load_frag(&B1[cur][wn*32 + ns*16][kk*32], 72, lane);
        v16bf b3f = load_frag(&B3[cur][wn*32 + ns*16][kk*32], 72, lane);
        #pragma unroll
        for (int ms = 0; ms < 2; ++ms) {
          a1[ms][ns] = wmma_bf16(af[ms], b1f, a1[ms][ns]);
          a3[ms][ns] = wmma_bf16(af[ms], b3f, a3[ms][ns]);
        }
      }
    }
    wait_async0();
    __syncthreads();
    cur ^= 1;
  }

  #pragma unroll
  for (int ms = 0; ms < 2; ++ms)
    #pragma unroll
    for (int ns = 0; ns < 2; ++ns) {
      int rbase = m0 + wm*32 + ms*16 + ((lane >> 4) * 8);
      int col   = n0 + wn*32 + ns*16 + (lane & 15);
      #pragma unroll
      for (int r = 0; r < 8; ++r) {
        float u = a1[ms][ns][r];
        float sig = fast_rcp(1.0f + __expf(-u));
        float g = u * sig * a3[ms][ns][r];
        G[(size_t)(rbase + r) * N + col] = (__bf16)g;
      }
    }
}

// ---------------------------------------------------------------------------
// RoPE + repack to [B,H,T,D] bf16 (q,k rotated; v copied).
// ---------------------------------------------------------------------------
__global__ void rope_pack(const __bf16* __restrict__ qb, const __bf16* __restrict__ kb,
                          const __bf16* __restrict__ vb,
                          const float* __restrict__ cosb, const float* __restrict__ sinb,
                          __bf16* __restrict__ qr, __bf16* __restrict__ kr,
                          __bf16* __restrict__ vr) {
  size_t idx = (size_t)blockIdx.x * blockDim.x + threadIdx.x;  // over B*T*H*D
  int d = idx & (DD - 1);
  size_t rst = idx >> 6;
  int h = rst & (HH - 1); rst >>= 4;
  int t = rst & (TT - 1);
  int b = (int)(rst >> 11);

  size_t src = ((size_t)b * TT + t) * CC + h * DD + d;
  size_t dst = (((size_t)b * HH + h) * TT + t) * DD + d;
  int dp = (d < 32) ? d + 32 : d - 32;
  float sgn = (d < 32) ? -1.f : 1.f;
  size_t srcp = src - d + dp;
  float cs = cosb[t * DD + d], sn = sinb[t * DD + d];

  qr[dst] = (__bf16)((float)qb[src] * cs + sgn * (float)qb[srcp] * sn);
  kr[dst] = (__bf16)((float)kb[src] * cs + sgn * (float)kb[srcp] * sn);
  vr[dst] = vb[src];
}

// ---------------------------------------------------------------------------
// Flash attention (causal). Block = 128 thr = 4 waves, each wave 16 queries.
// grid = (T/64, H, B). Q*K^T and P*V through WMMA; online softmax with
// width-16 shfl_xor row reductions matching the C-matrix layout.
// K tile staged with async global->LDS; V staged transposed via ds stores.
// ---------------------------------------------------------------------------
__global__ void flash_attn(const __bf16* __restrict__ q, const __bf16* __restrict__ k,
                           const __bf16* __restrict__ v, __bf16* __restrict__ y) {
  __shared__ __align__(16) __bf16 Ks[32][64];      // [key][dim]
  __shared__ __align__(16) __bf16 Vt[64][32];      // [dim][key] (transposed)
  __shared__ __align__(16) __bf16 Ps[4][16][32];   // per-wave P staging

  int tid  = threadIdx.x;
  int lane = tid & 31;
  int wave = tid >> 5;
  int l15 = lane & 15;
  int g8  = (lane >> 4) * 8;

  int hh = blockIdx.y;
  int b  = blockIdx.z;
  int qbase_wg = blockIdx.x * 64;
  int qbase = qbase_wg + wave * 16;
  size_t headbase = ((size_t)b * HH + hh) * TT;

  // Q fragments (A-matrix layout), head-dim 64 -> two K=32 fragments
  v16bf qf[2];
  {
    const __bf16* qp = &q[(headbase + qbase + l15) * DD];
    #pragma unroll
    for (int kk = 0; kk < 2; ++kk) {
      v8bf lo = *(const v8bf*)(qp + kk*32 + g8);
      v8bf hi = *(const v8bf*)(qp + kk*32 + 16 + g8);
      qf[kk] = frag_combine(lo, hi);
    }
  }

  v8f o[4]; float mstate[8], lstate[8];
  #pragma unroll
  for (int i = 0; i < 4; ++i) o[i] = vzero8();
  #pragma unroll
  for (int r = 0; r < 8; ++r) { mstate[r] = -1e30f; lstate[r] = 0.f; }

  const void* kbase_tile = (const void*)&k[headbase * DD];

  int nkb = blockIdx.x * 2 + 2;      // key tiles of 32 covering keys < qbase_wg+64
  for (int kb = 0; kb < nkb; ++kb) {
    int keybase = kb * 32;
    // stage K tile [32][64] via async global->LDS (256 16B chunks, 2/thread)
    #pragma unroll
    for (int tch = 0; tch < 2; ++tch) {
      int c = tid + tch * 128;
      int row = c >> 3, koff = (c & 7) * 8;
      async_b128(&Ks[row][koff],
                 (unsigned)(((keybase + row) * DD + koff) * 2), kbase_tile);
    }
    // stage V transposed [64][32]
    #pragma unroll
    for (int tch = 0; tch < 2; ++tch) {
      int c = tid + tch * 128;
      int key = c >> 3, dof = (c & 7) * 8;
      v8bf vv = *(const v8bf*)&v[(headbase + keybase + key) * DD + dof];
      #pragma unroll
      for (int i = 0; i < 8; ++i) Vt[dof + i][key] = vv[i];
    }
    wait_async0();
    __syncthreads();

    bool active = (keybase <= qbase + 15);
    if (active) {
      // S = Q*K^T (16 queries x 32 keys)
      v8f s[2]; s[0] = vzero8(); s[1] = vzero8();
      #pragma unroll
      for (int ns = 0; ns < 2; ++ns)
        #pragma unroll
        for (int kk = 0; kk < 2; ++kk) {
          const __bf16* p = &Ks[ns*16 + l15][kk*32];
          v16bf bf = frag_combine(*(const v8bf*)(p + g8), *(const v8bf*)(p + 16 + g8));
          s[ns] = wmma_bf16(qf[kk], bf, s[ns]);
        }
      // scale + causal mask
      int qrow0 = qbase + (lane >> 4) * 8;
      #pragma unroll
      for (int ns = 0; ns < 2; ++ns)
        #pragma unroll
        for (int r = 0; r < 8; ++r) {
          int key = keybase + ns*16 + l15;
          float val = s[ns][r] * 0.125f;
          s[ns][r] = (key <= qrow0 + r) ? val : -1e30f;
        }
      // online softmax: row stats via width-16 butterflies
      float alpha[8];
      #pragma unroll
      for (int r = 0; r < 8; ++r) {
        float rm = fmaxf(s[0][r], s[1][r]);
        #pragma unroll
        for (int m = 8; m >= 1; m >>= 1) rm = fmaxf(rm, __shfl_xor(rm, m, 16));
        float mnew = fmaxf(mstate[r], rm);
        alpha[r] = __expf(mstate[r] - mnew);
        mstate[r] = mnew;
      }
      #pragma unroll
      for (int ns = 0; ns < 2; ++ns)
        #pragma unroll
        for (int r = 0; r < 8; ++r) s[ns][r] = __expf(s[ns][r] - mstate[r]);
      #pragma unroll
      for (int r = 0; r < 8; ++r) {
        float rs = s[0][r] + s[1][r];
        #pragma unroll
        for (int m = 8; m >= 1; m >>= 1) rs += __shfl_xor(rs, m, 16);
        lstate[r] = lstate[r] * alpha[r] + rs;
      }
      #pragma unroll
      for (int nd = 0; nd < 4; ++nd)
        #pragma unroll
        for (int r = 0; r < 8; ++r) o[nd][r] *= alpha[r];
      // stage P (C-layout -> LDS -> A-layout); wave-private, same-wave LDS order
      #pragma unroll
      for (int ns = 0; ns < 2; ++ns)
        #pragma unroll
        for (int r = 0; r < 8; ++r)
          Ps[wave][(lane >> 4) * 8 + r][ns*16 + l15] = (__bf16)s[ns][r];
      const __bf16* pp = &Ps[wave][l15][0];
      v16bf pf = frag_combine(*(const v8bf*)(pp + g8), *(const v8bf*)(pp + 16 + g8));
      // O += P*V
      #pragma unroll
      for (int nd = 0; nd < 4; ++nd) {
        const __bf16* vp = &Vt[nd*16 + l15][0];
        v16bf vf = frag_combine(*(const v8bf*)(vp + g8), *(const v8bf*)(vp + 16 + g8));
        o[nd] = wmma_bf16(pf, vf, o[nd]);
      }
    }
    __syncthreads();
  }

  // normalize + store y as [B,T,H*D] bf16 (GEMM-ready for O-projection)
  #pragma unroll
  for (int r = 0; r < 8; ++r) {
    float inv = fast_rcp(lstate[r]);
    #pragma unroll
    for (int nd = 0; nd < 4; ++nd) {
      int qq = qbase + (lane >> 4) * 8 + r;
      float val = o[nd][r] * inv;
      y[((size_t)b * TT + qq) * CC + hh * DD + nd*16 + l15] = (__bf16)val;
    }
  }
}

// ---------------------------------------------------------------------------
// Router: per-token logits over 8 experts, top-2 softmax gate.
// ---------------------------------------------------------------------------
__global__ void router_topk(const __bf16* __restrict__ hn, const float* __restrict__ rw,
                            float* __restrict__ gate) {
  __shared__ float red[256][EE];
  int t = blockIdx.x, tid = threadIdx.x;
  float acc[EE];
  #pragma unroll
  for (int e = 0; e < EE; ++e) acc[e] = 0.f;
  for (int c = tid; c < CC; c += 256) {
    float xv = (float)hn[(size_t)t * CC + c];
    #pragma unroll
    for (int e = 0; e < EE; ++e) acc[e] += xv * rw[(size_t)e * CC + c];
  }
  #pragma unroll
  for (int e = 0; e < EE; ++e) red[tid][e] = acc[e];
  __syncthreads();
  for (int s = 128; s > 0; s >>= 1) {
    if (tid < s)
      #pragma unroll
      for (int e = 0; e < EE; ++e) red[tid][e] += red[tid + s][e];
    __syncthreads();
  }
  if (tid == 0) {
    float lg[EE];
    #pragma unroll
    for (int e = 0; e < EE; ++e) lg[e] = red[0][e];
    int i1 = 0;
    for (int e = 1; e < EE; ++e) if (lg[e] > lg[i1]) i1 = e;
    int i2 = -1;
    for (int e = 0; e < EE; ++e) if (e != i1 && (i2 < 0 || lg[e] > lg[i2])) i2 = e;
    float e2 = __expf(lg[i2] - lg[i1]);
    float inv = fast_rcp(1.0f + e2);
    #pragma unroll
    for (int e = 0; e < EE; ++e) gate[(size_t)t * EE + e] = 0.f;
    gate[(size_t)t * EE + i1] = inv;
    gate[(size_t)t * EE + i2] = e2 * inv;
  }
}

// ---------------------------------------------------------------------------
// Host-side orchestration
// ---------------------------------------------------------------------------
extern "C" void kernel_launch(void* const* d_in, const int* in_sizes, int n_in,
                              void* d_out, int out_size, void* d_ws, size_t ws_size,
                              hipStream_t stream) {
  const float* x         = (const float*)d_in[0];
  const float* rope_cos  = (const float*)d_in[1];
  const float* rope_sin  = (const float*)d_in[2];
  const float* attn_norm = (const float*)d_in[3];
  const float* q_w       = (const float*)d_in[4];
  const float* k_w       = (const float*)d_in[5];
  const float* v_w       = (const float*)d_in[6];
  const float* o_w       = (const float*)d_in[7];
  const float* ffn_norm  = (const float*)d_in[8];
  const float* router_w  = (const float*)d_in[9];
  const float* w1        = (const float*)d_in[10];
  const float* w2        = (const float*)d_in[11];
  const float* w3        = (const float*)d_in[12];
  float* out = (float*)d_out;

  // carve workspace
  size_t off = 0;
  auto alloc = [&](size_t bytes) -> void* {
    void* p = (char*)d_ws + off;
    off += (bytes + 255) & ~(size_t)255;
    return p;
  };
  const size_t wCC  = (size_t)CC * CC;
  const size_t wEHC = (size_t)EE * HDIM * CC;
  const size_t tok  = (size_t)MM * CC;

  __bf16* hq   = (__bf16*)alloc(wCC * 2);
  __bf16* hk   = (__bf16*)alloc(wCC * 2);
  __bf16* hv   = (__bf16*)alloc(wCC * 2);
  __bf16* ho   = (__bf16*)alloc(wCC * 2);
  __bf16* hw1  = (__bf16*)alloc(wEHC * 2);
  __bf16* hw2  = (__bf16*)alloc(wEHC * 2);
  __bf16* hw3  = (__bf16*)alloc(wEHC * 2);
  __bf16* h_bf = (__bf16*)alloc(tok * 2);
  __bf16* qb   = (__bf16*)alloc(tok * 2);
  __bf16* kb   = (__bf16*)alloc(tok * 2);
  __bf16* vb   = (__bf16*)alloc(tok * 2);
  __bf16* qr   = (__bf16*)alloc(tok * 2);
  __bf16* kr   = (__bf16*)alloc(tok * 2);
  __bf16* vr   = (__bf16*)alloc(tok * 2);
  __bf16* y_bf = (__bf16*)alloc(tok * 2);
  float*  xmid = (float*) alloc(tok * 4);
  __bf16* hn   = (__bf16*)alloc(tok * 2);
  float*  gate = (float*) alloc((size_t)MM * EE * 4);
  __bf16* g_bf = (__bf16*)alloc((size_t)MM * HDIM * 2);

  // 1) weight casts fp32 -> bf16
  castf2b<<<4096, 256, 0, stream>>>(q_w, hq, wCC);
  castf2b<<<4096, 256, 0, stream>>>(k_w, hk, wCC);
  castf2b<<<4096, 256, 0, stream>>>(v_w, hv, wCC);
  castf2b<<<4096, 256, 0, stream>>>(o_w, ho, wCC);
  castf2b<<<8192, 256, 0, stream>>>(w1, hw1, wEHC);
  castf2b<<<8192, 256, 0, stream>>>(w2, hw2, wEHC);
  castf2b<<<8192, 256, 0, stream>>>(w3, hw3, wEHC);

  // 2) attn RMSNorm
  rmsnorm_cast<<<MM, 256, 0, stream>>>(x, attn_norm, h_bf);

  // 3) QKV projections (WMMA)
  dim3 gg(CC / 64, MM / 128);
  gemm_wmma<0><<<gg, 256, 0, stream>>>(h_bf, hq, MM, CC, CC, qb, nullptr, nullptr, nullptr, nullptr, 0);
  gemm_wmma<0><<<gg, 256, 0, stream>>>(h_bf, hk, MM, CC, CC, kb, nullptr, nullptr, nullptr, nullptr, 0);
  gemm_wmma<0><<<gg, 256, 0, stream>>>(h_bf, hv, MM, CC, CC, vb, nullptr, nullptr, nullptr, nullptr, 0);

  // 4) RoPE + [B,H,T,D] repack
  rope_pack<<<(MM * CC) / 256, 256, 0, stream>>>(qb, kb, vb, rope_cos, rope_sin, qr, kr, vr);

  // 5) causal flash attention (WMMA)
  dim3 ag(TT / 64, HH, BB);
  flash_attn<<<ag, 128, 0, stream>>>(qr, kr, vr, y_bf);

  // 6) O-projection + residual (writes xmid and seeds d_out)
  gemm_wmma<1><<<gg, 256, 0, stream>>>(y_bf, ho, MM, CC, CC, nullptr, xmid, out, x, nullptr, 0);

  // 7) ffn RMSNorm
  rmsnorm_cast<<<MM, 256, 0, stream>>>(xmid, ffn_norm, hn);

  // 8) router + top-2 gate
  router_topk<<<MM, 256, 0, stream>>>(hn, router_w, gate);

  // 9) MoE: dense expert loop, gated accumulation into d_out
  dim3 g13(HDIM / 64, MM / 128);
  for (int e = 0; e < EE; ++e) {
    moe13_wmma<<<g13, 256, 0, stream>>>(hn, hw1 + (size_t)e * HDIM * CC,
                                        hw3 + (size_t)e * HDIM * CC, g_bf, CC, HDIM);
    gemm_wmma<2><<<gg, 256, 0, stream>>>(g_bf, hw2 + (size_t)e * CC * HDIM,
                                         MM, CC, HDIM, nullptr, out, nullptr, nullptr, gate, e);
  }
  (void)in_sizes; (void)n_in; (void)out_size; (void)ws_size;
}
